// HAN_lp_37452114821483
// MI455X (gfx1250) — matrix operand order repaired and checked by reading, not compiled, not run on hardware
//
#include <hip/hip_runtime.h>
#include <hip/hip_bf16.h>

// ---------------- problem constants (match reference) ----------------
#define NN   20000   // nodes
#define INF_ 256     // input feature dim
#define HH   8       // heads
#define DD   32      // per-head dim
#define KK   256     // H*D
#define OUTD 64      // final embedding
#define EE   256000  // edges per metapath graph
#define PP   100000  // pos/neg pairs
#define SEMD 128     // semantic hidden
#define SLOPE 0.2f

typedef __attribute__((ext_vector_type(16))) __bf16 v16bf;
typedef __attribute__((ext_vector_type(8)))  __bf16 v8bf;
typedef __attribute__((ext_vector_type(8)))  float  v8f;

// ---------------------------------------------------------------------
// LDS-staged tiled GEMM: C[M,Nc] = A[M,Kd] @ B[Kd,Nc] (+ bias[Nc]).
// fp32 in memory, bf16 WMMA with fp32 accumulation.
// Block = 256 threads = 8 waves (2 M-tiles x 4 N-tiles) -> 32x64 C tile.
// Per K-step (32):
//   * A slab 32x32 f32 -> LDS via global_load_async_to_lds_b128 (ASYNCcnt)
//   * B slab 32x64 f32 -> coalesced float4 loads, cvt to bf16, ds_store
//     transposed into Bt[64][40] (80B rows => 16B-aligned wave reads)
//   * each wave: 4x float4 + 2x v8bf LDS reads, one v_wmma_f32_16x16x32_bf16
// Requires M%32==0, Nc%64==0, Kd%32==0  (20000/256/128/64 all satisfy).
// ---------------------------------------------------------------------
__global__ void gemm_bf16_wmma(const float* __restrict__ A,
                               const float* __restrict__ B,
                               const float* __restrict__ bias,
                               float* __restrict__ C,
                               int M, int Kd, int Nc) {
    __shared__ __align__(16) float  As[32][32];   // 4 KB, async-filled (f32)
    __shared__ __align__(16) __bf16 Bt[64][40];   // 5 KB, transposed bf16, padded

    const int tid  = threadIdx.x;
    const int lane = tid & 31;
    const int wid  = tid >> 5;          // 0..7
    const int wm   = wid >> 2;          // 0..1  (16-row sub-tile)
    const int wn   = wid & 3;           // 0..3  (16-col sub-tile)

    const int tilesN = Nc >> 6;         // 64-col block tiles
    const int m0 = (blockIdx.x / tilesN) * 32;
    const int n0 = (blockIdx.x % tilesN) * 64;

    // staging coordinates
    const int arow = tid >> 3;          // 0..31
    const int acol = (tid & 7) * 4;     // 0,4,..,28  (16B per lane)
    const unsigned a_lds = (unsigned)(uintptr_t)&As[arow][acol]; // LDS byte addr = addr[31:0]

    const int brow = tid >> 3;          // 0..31 (K row)
    const int bcol = (tid & 7) * 8;     // 0..56 (8 cols per thread)

    // wave operand coordinates (ISA 7.12.2 wave32 layouts)
    const int arow_l = wm * 16 + (lane & 15);
    const int kb     = (lane < 16) ? 0 : 8;
    const int bcol_l = wn * 16 + (lane & 15);
    const int bk     = (lane < 16) ? 0 : 8;

    v8f acc = {};
    for (int k0 = 0; k0 < Kd; k0 += 32) {
        if (k0) __syncthreads();        // WAR: previous tile fully consumed

        // ---- A: async copy 16B/lane global -> LDS (raw f32) ----
        {
            uint64_t ga = (uint64_t)(A + (size_t)(m0 + arow) * Kd + k0 + acol);
            asm volatile("global_load_async_to_lds_b128 %0, %1, off"
                         :: "v"(a_lds), "v"(ga) : "memory");
        }

        // ---- B: coalesced load, cvt, transposed ds_store ----
        {
            const float* gb = B + (size_t)(k0 + brow) * Nc + n0 + bcol;
            float4 b0 = *(const float4*)(gb);
            float4 b1 = *(const float4*)(gb + 4);
            Bt[bcol + 0][brow] = (__bf16)b0.x;
            Bt[bcol + 1][brow] = (__bf16)b0.y;
            Bt[bcol + 2][brow] = (__bf16)b0.z;
            Bt[bcol + 3][brow] = (__bf16)b0.w;
            Bt[bcol + 4][brow] = (__bf16)b1.x;
            Bt[bcol + 5][brow] = (__bf16)b1.y;
            Bt[bcol + 6][brow] = (__bf16)b1.z;
            Bt[bcol + 7][brow] = (__bf16)b1.w;
        }

        asm volatile("s_wait_asynccnt 0x0" ::: "memory");
        __syncthreads();

        // ---- pack A operand: a[j] j<8 -> K=kb+j ; j>=8 -> K=kb+16+(j-8) ----
        float4 a0 = *(const float4*)&As[arow_l][kb];
        float4 a1 = *(const float4*)&As[arow_l][kb + 4];
        float4 a2 = *(const float4*)&As[arow_l][kb + 16];
        float4 a3 = *(const float4*)&As[arow_l][kb + 20];
        v16bf a;
        a[0]  = (__bf16)a0.x; a[1]  = (__bf16)a0.y; a[2]  = (__bf16)a0.z; a[3]  = (__bf16)a0.w;
        a[4]  = (__bf16)a1.x; a[5]  = (__bf16)a1.y; a[6]  = (__bf16)a1.z; a[7]  = (__bf16)a1.w;
        a[8]  = (__bf16)a2.x; a[9]  = (__bf16)a2.y; a[10] = (__bf16)a2.z; a[11] = (__bf16)a2.w;
        a[12] = (__bf16)a3.x; a[13] = (__bf16)a3.y; a[14] = (__bf16)a3.z; a[15] = (__bf16)a3.w;

        // ---- pack B operand: col=lane&15, b[j] j<8 -> K=bk+j ; j>=8 -> K=bk+16+(j-8) ----
        v8bf bl = *(const v8bf*)&Bt[bcol_l][bk];
        v8bf bh = *(const v8bf*)&Bt[bcol_l][bk + 16];
        v16bf b = __builtin_shufflevector(bl, bh, 0,1,2,3,4,5,6,7,8,9,10,11,12,13,14,15);

        acc = __builtin_amdgcn_wmma_f32_16x16x32_bf16(
            false, a, false, b, (short)0, acc, false, false);
    }

    // C/D layout: VGPR r -> row +r (+8 for lanes>=16), col = lane&15
    const int crow0 = m0 + wm * 16 + ((lane < 16) ? 0 : 8);
    const int ccol  = n0 + wn * 16 + (lane & 15);
    const float bv  = bias ? bias[ccol] : 0.0f;
    #pragma unroll
    for (int r = 0; r < 8; ++r)
        C[(size_t)(crow0 + r) * Nc + ccol] = acc[r] + bv;
}

// ---------------------------------------------------------------------
__global__ void fill_f32(float* __restrict__ p, float v, int n) {
    int i = blockIdx.x * blockDim.x + threadIdx.x;
    if (i < n) p[i] = v;
}

// el[n,h] = sum_d feat[n,h,d]*al[h,d]; er likewise
__global__ void attn_logits(const float* __restrict__ feat,
                            const float* __restrict__ al,
                            const float* __restrict__ ar,
                            float* __restrict__ el, float* __restrict__ er) {
    int t = blockIdx.x * blockDim.x + threadIdx.x;
    if (t >= NN * HH) return;
    int n = t / HH, h = t - n * HH;
    const float* f   = feat + (size_t)n * KK + h * DD;
    const float* alh = al + h * DD;
    const float* arh = ar + h * DD;
    float sl = 0.f, sr = 0.f;
    #pragma unroll 8
    for (int d = 0; d < DD; ++d) { sl += f[d] * alh[d]; sr += f[d] * arh[d]; }
    el[t] = sl; er[t] = sr;
}

// e = leaky_relu(el[src]+er[dst]); running max per (dst,h) via f32 atomic max
__global__ void edge_logit_max(const int* __restrict__ src, const int* __restrict__ dst,
                               const float* __restrict__ el, const float* __restrict__ er,
                               float* __restrict__ ebuf, float* __restrict__ mbuf) {
    int t = blockIdx.x * blockDim.x + threadIdx.x;
    if (t >= EE * HH) return;
    int e = t >> 3, h = t & 7;               // HH == 8
    int s = src[e], d = dst[e];
    float v = el[s * HH + h] + er[d * HH + h];
    v = (v > 0.f) ? v : SLOPE * v;
    ebuf[t] = v;
    atomicMax(&mbuf[d * HH + h], v);         // -> global_atomic_max_num_f32
}

// ex = exp(e - m[dst]); running sum per (dst,h)
__global__ void edge_exp_sum(const int* __restrict__ dst,
                             float* __restrict__ ebuf,
                             const float* __restrict__ mbuf,
                             float* __restrict__ sbuf) {
    int t = blockIdx.x * blockDim.x + threadIdx.x;
    if (t >= EE * HH) return;
    int e = t >> 3, h = t & 7;
    int d = dst[e];
    float ex = __expf(ebuf[t] - mbuf[d * HH + h]);
    ebuf[t] = ex;
    atomicAdd(&sbuf[d * HH + h], ex);
}

// agg[dst,h,:] += feat[src,h,:] * (ex / s[dst,h]); one thread per (edge,head)
__global__ void edge_aggregate(const int* __restrict__ src, const int* __restrict__ dst,
                               const float* __restrict__ ebuf, const float* __restrict__ sbuf,
                               const float* __restrict__ feat, float* __restrict__ agg) {
    int t = blockIdx.x * blockDim.x + threadIdx.x;
    if (t >= EE * HH) return;
    int e = t >> 3, h = t & 7;
    int s = src[e], d = dst[e];
    float alpha = ebuf[t] / sbuf[d * HH + h];
    const float* f = feat + (size_t)s * KK + h * DD;
    float*       o = agg  + (size_t)d * KK + h * DD;
    #pragma unroll 8
    for (int dd = 0; dd < DD; ++dd) atomicAdd(&o[dd], f[dd] * alpha);
}

// emb = elu(agg + bg)
__global__ void elu_bias(const float* __restrict__ agg, const float* __restrict__ bg,
                         float* __restrict__ emb) {
    int i = blockIdx.x * blockDim.x + threadIdx.x;
    if (i >= NN * KK) return;
    float v = agg[i] + bg[i & (KK - 1)];
    emb[i] = (v > 0.f) ? v : (__expf(v) - 1.0f);
}

// acc += sum_n dot(tanh(tproj[n,:]+bs1), Ws2)   (mean applied later)
__global__ void sem_reduce(const float* __restrict__ tproj, const float* __restrict__ bs1,
                           const float* __restrict__ ws2, float* __restrict__ acc) {
    int n = blockIdx.x * blockDim.x + threadIdx.x;
    if (n >= NN) return;
    const float* tp = tproj + (size_t)n * SEMD;
    float v = 0.f;
    #pragma unroll 8
    for (int j = 0; j < SEMD; ++j) v += tanhf(tp[j] + bs1[j]) * ws2[j];
    atomicAdd(acc, v);
}

// h = softmax(wacc/N)-weighted combo of emb0, emb1 (written in place over emb0)
__global__ void sem_combine(const float* __restrict__ wacc,
                            const float* __restrict__ emb0, const float* __restrict__ emb1,
                            float* __restrict__ h) {
    int i = blockIdx.x * blockDim.x + threadIdx.x;
    if (i >= NN * KK) return;
    float w0 = wacc[0] * (1.0f / NN), w1 = wacc[1] * (1.0f / NN);
    float mx = fmaxf(w0, w1);
    float e0 = __expf(w0 - mx), e1 = __expf(w1 - mx);
    float b0 = e0 / (e0 + e1);
    h[i] = b0 * emb0[i] + (1.0f - b0) * emb1[i];
}

// DistMult scoring: out[p] = sum_k hu[i,k]*r[k]*hv[j,k]
__global__ void score_edges(const float* __restrict__ hu, const float* __restrict__ hv,
                            const float* __restrict__ r, const int* __restrict__ edges,
                            float* __restrict__ out) {
    int p = blockIdx.x * blockDim.x + threadIdx.x;
    if (p >= PP) return;
    int i = edges[2 * p], j = edges[2 * p + 1];
    const float* a = hu + (size_t)i * OUTD;
    const float* b = hv + (size_t)j * OUTD;
    float s = 0.f;
    #pragma unroll 8
    for (int k = 0; k < OUTD; ++k) s += a[k] * r[k] * b[k];
    out[p] = s;
}

// ---------------------------------------------------------------------
static inline int cdiv(long long a, int b) { return (int)((a + b - 1) / b); }

extern "C" void kernel_launch(void* const* d_in, const int* in_sizes, int n_in,
                              void* d_out, int out_size, void* d_ws, size_t ws_size,
                              hipStream_t stream) {
    (void)in_sizes; (void)n_in; (void)out_size; (void)ws_size;

    // ---- workspace carve-out (floats) ----
    float* W = (float*)d_ws;
    size_t off = 0;
    float* emb[4]; // u0,u1,v0,v1
    for (int i = 0; i < 4; ++i) { emb[i] = W + off; off += (size_t)NN * KK; }
    float* feat = W + off; off += (size_t)NN * KK;   // also reused as tproj [N,SEM]
    float* agg  = W + off; off += (size_t)NN * KK;
    float* el   = W + off; off += (size_t)NN * HH;
    float* er   = W + off; off += (size_t)NN * HH;
    float* mbuf = W + off; off += (size_t)NN * HH;
    float* sbuf = W + off; off += (size_t)NN * HH;
    float* ebuf = W + off; off += (size_t)EE * HH;
    float* wacc = W + off; off += 8;                 // [u0,u1,v0,v1,...]
    float* hfin[2];
    for (int i = 0; i < 2; ++i) { hfin[i] = W + off; off += (size_t)NN * OUTD; }

    const int BT = 256;
    auto gemm = [&](const float* A, const float* B, const float* bias, float* C,
                    int M, int Kd, int Nc) {
        int blocks = (M / 32) * (Nc / 64);           // 32x64 block tiles
        gemm_bf16_wmma<<<blocks, BT, 0, stream>>>(A, B, bias, C, M, Kd, Nc);
    };

    fill_f32<<<1, 8, 0, stream>>>(wacc, 0.0f, 8);

    for (int s = 0; s < 2; ++s) {
        const float* x = (const float*)d_in[s];
        const int base = 2 + s * 17;
        for (int m = 0; m < 2; ++m) {
            const int*   srcI = (const int*)  d_in[base + m * 6 + 0];
            const int*   dstI = (const int*)  d_in[base + m * 6 + 1];
            const float* Wg   = (const float*)d_in[base + m * 6 + 2];
            const float* al_  = (const float*)d_in[base + m * 6 + 3];
            const float* ar_  = (const float*)d_in[base + m * 6 + 4];
            const float* bg   = (const float*)d_in[base + m * 6 + 5];

            fill_f32<<<cdiv((size_t)NN * KK, BT), BT, 0, stream>>>(agg,  0.0f,      NN * KK);
            fill_f32<<<cdiv(NN * HH, BT),          BT, 0, stream>>>(mbuf, -3.0e38f, NN * HH);
            fill_f32<<<cdiv(NN * HH, BT),          BT, 0, stream>>>(sbuf, 0.0f,     NN * HH);

            gemm(x, Wg, nullptr, feat, NN, INF_, KK);                       // feat = x @ Wg
            attn_logits<<<cdiv(NN * HH, BT), BT, 0, stream>>>(feat, al_, ar_, el, er);
            edge_logit_max<<<cdiv((size_t)EE * HH, BT), BT, 0, stream>>>(srcI, dstI, el, er, ebuf, mbuf);
            edge_exp_sum  <<<cdiv((size_t)EE * HH, BT), BT, 0, stream>>>(dstI, ebuf, mbuf, sbuf);
            edge_aggregate<<<cdiv((size_t)EE * HH, BT), BT, 0, stream>>>(srcI, dstI, ebuf, sbuf, feat, agg);
            elu_bias<<<cdiv((size_t)NN * KK, BT), BT, 0, stream>>>(agg, bg, emb[s * 2 + m]);
        }

        const float* Ws1 = (const float*)d_in[base + 12];
        const float* bs1 = (const float*)d_in[base + 13];
        const float* Ws2 = (const float*)d_in[base + 14];
        const float* Wp  = (const float*)d_in[base + 15];
        const float* bp  = (const float*)d_in[base + 16];

        for (int m = 0; m < 2; ++m) {
            gemm(emb[s * 2 + m], Ws1, nullptr, feat, NN, KK, SEMD);        // tproj = emb @ Ws1
            sem_reduce<<<cdiv(NN, BT), BT, 0, stream>>>(feat, bs1, Ws2, wacc + s * 2 + m);
        }
        sem_combine<<<cdiv((size_t)NN * KK, BT), BT, 0, stream>>>(
            wacc + s * 2, emb[s * 2 + 0], emb[s * 2 + 1], emb[s * 2 + 0]); // h (in place)
        gemm(emb[s * 2 + 0], Wp, bp, hfin[s], NN, KK, OUTD);               // hfin = h @ Wp + bp
    }

    const float* r   = (const float*)d_in[36];
    const int*   pos = (const int*)  d_in[37];
    const int*   neg = (const int*)  d_in[38];
    float* out = (float*)d_out;
    score_edges<<<cdiv(PP, BT), BT, 0, stream>>>(hfin[0], hfin[1], r, pos, out);
    score_edges<<<cdiv(PP, BT), BT, 0, stream>>>(hfin[0], hfin[1], r, neg, out + PP);
}
